// ViSNetBlock_31602369364622
// MI455X (gfx1250) — compile-verified, matrix operand order checked by reading, not run on hardware
//
#include <hip/hip_runtime.h>
#include <math.h>

// ---------------- problem constants ----------------
#define N_NODES 6000
#define N_EDGES 192000
#define CC      128
#define NC      (N_NODES*CC)      // 768000
#define N3C     (N_NODES*3*CC)    // 2304000
#define NODE_TILES (N_NODES/16)   // 375
#define EDGE_TILES (N_EDGES/16)   // 12000
#define PI_F 3.14159265358979323846f

// bf16 weight region offsets (u16 elements) inside workspace
#define OWQ    0
#define OWK    16384
#define OWV    32768
#define OWVEC  49152      // 3C x C = 49152 (vec1 | vec2 | vec3 blocks of 16384)
#define OWTRG  98304
#define OWSRC  114688
#define OWVD   131072
#define OWDK   147456
#define OWDV   163840
#define OWS    180224     // 2C x C = 32768
#define OWF    212992     // 2C x C
#define OWWD   245760
#define OWO    262144     // 3C x C (o1 | o2 | o3 blocks of 16384)
#define WBF_TOTAL 311296  // u16 elems -> 622592 bytes
// required ws: 622592 B + 15,360,000 f32 ~= 62 MB

typedef __attribute__((ext_vector_type(16))) __bf16 v16bf;
typedef __attribute__((ext_vector_type(8)))  float  v8f;
typedef unsigned int u32a __attribute__((may_alias));

union ABfrag { v16bf v; unsigned int u[8]; };

__device__ __forceinline__ unsigned short f2bf(float x) {
    unsigned int u = __float_as_uint(x);
    u += 0x7FFFu + ((u >> 16) & 1u);        // round to nearest even
    return (unsigned short)(u >> 16);
}
__device__ __forceinline__ float bf2f(unsigned short b) {
    return __uint_as_float((unsigned int)b << 16);
}
// pack two bf16 into one LDS dword store
__device__ __forceinline__ void st2bf(unsigned short* p, float a, float b) {
    *(u32a*)p = (unsigned int)f2bf(a) | ((unsigned int)f2bf(b) << 16);
}
__device__ __forceinline__ float siluf(float x) { return x / (1.f + __expf(-x)); }

__device__ __forceinline__ float wavesum(float x) {
    x += __shfl_xor(x, 1, 32);  x += __shfl_xor(x, 2, 32);
    x += __shfl_xor(x, 4, 32);  x += __shfl_xor(x, 8, 32);
    x += __shfl_xor(x, 16, 32);
    return x;
}
__device__ __forceinline__ float rowsum16(float x) {   // sum across 16-lane half
    x += __shfl_xor(x, 1, 32);  x += __shfl_xor(x, 2, 32);
    x += __shfl_xor(x, 4, 32);  x += __shfl_xor(x, 8, 32);
    return x;
}

// A fragment (16x32 bf16 tile, M across lanes) from LDS row-major [16][CC] ushort
__device__ __forceinline__ v16bf loadA(const unsigned short* tile, int kt, int lm, int lgrp) {
    ABfrag f;
    const int kbase = kt * 32 + 8 * lgrp;
    const unsigned short* row = tile + lm * CC;
#pragma unroll
    for (int j = 0; j < 8; ++j) {
        int k = kbase + ((j < 4) ? 0 : 16) + 2 * (j & 3);
        f.u[j] = *(const u32a*)(row + k);
    }
    return f.v;
}
// B fragment (32x16, B[k,n] = W[n,k]) from bf16 weights row-major [outdim][CC]
__device__ __forceinline__ v16bf loadB(const unsigned short* W, int nt, int kt, int lm, int lgrp) {
    ABfrag f;
    const int kbase = kt * 32 + 8 * lgrp;
    const unsigned short* row = W + (size_t)(nt * 16 + lm) * CC;
#pragma unroll
    for (int j = 0; j < 8; ++j) {
        int k = kbase + ((j < 4) ? 0 : 16) + 2 * (j & 3);
        f.u[j] = *(const u32a*)(row + k);
    }
    return f.v;
}

// preloaded A fragments for a 16xCC tile (reused across several GEMMs)
struct AF { v16bf a[4]; };
__device__ __forceinline__ void loadAF(const unsigned short* tile, AF& A, int lm, int lgrp) {
#pragma unroll
    for (int kt = 0; kt < 4; ++kt) A.a[kt] = loadA(tile, kt, lm, lgrp);
}

// GEMM with register-resident A fragments (for tiles reused by several GEMMs)
template <int NT>
__device__ __forceinline__ void gemmR(const AF& A, const unsigned short* W,
                                      v8f* acc, int lm, int lgrp) {
    constexpr int TOT = 4 * NT;
    v16bf b0 = loadB(W, 0, 0, lm, lgrp);
    v16bf b1 = loadB(W, 1 % NT, 1 / NT, lm, lgrp);
#pragma unroll
    for (int i = 0; i < TOT; ++i) {
        const int kt = i / NT, nt = i % NT;
        v16bf bn = b1;
        if (i + 2 < TOT) bn = loadB(W, (i + 2) % NT, (i + 2) / NT, lm, lgrp);
        acc[nt] = __builtin_amdgcn_wmma_f32_16x16x32_bf16(
            false, A.a[kt], false, b0, (short)0, acc[nt], false, false);
        b0 = b1; b1 = bn;
    }
}

// GEMM with A read per k-tile from LDS (low register pressure variant)
template <int NT>
__device__ __forceinline__ void gemmL(const unsigned short* tileA, const unsigned short* W,
                                      v8f* acc, int lm, int lgrp) {
#pragma unroll
    for (int kt = 0; kt < 4; ++kt) {
        v16bf a = loadA(tileA, kt, lm, lgrp);
        v16bf b0 = loadB(W, 0, kt, lm, lgrp);
        v16bf b1 = (NT > 1) ? loadB(W, 1, kt, lm, lgrp) : b0;
#pragma unroll
        for (int nt = 0; nt < NT; ++nt) {
            v16bf bn = b1;
            if (nt + 2 < NT) bn = loadB(W, nt + 2, kt, lm, lgrp);
            acc[nt] = __builtin_amdgcn_wmma_f32_16x16x32_bf16(
                false, a, false, b0, (short)0, acc[nt], false, false);
            b0 = b1; b1 = bn;
        }
    }
}

__device__ __forceinline__ v8f zero8() {
    v8f z = {0.f, 0.f, 0.f, 0.f, 0.f, 0.f, 0.f, 0.f};
    return z;
}

// ---------------- small utility kernels ----------------
__global__ void cvt_bf16_kernel(const float* __restrict__ src, unsigned short* __restrict__ dst, int n) {
    int i = blockIdx.x * 256 + threadIdx.x;
    if (i < n) dst[i] = f2bf(src[i]);
}
__global__ void zerof_kernel(float* __restrict__ p, int n) {
    int i = blockIdx.x * 256 + threadIdx.x;
    if (i < n) p[i] = 0.f;
}

// ---------------- node stage ----------------
// min 4 waves/SIMD -> cap 256 VGPRs/wave (no MSB addressing, better latency hiding)
__global__ __launch_bounds__(256, 4) void node_stage(
    const float* __restrict__ x, const float* __restrict__ vec,
    const float* __restrict__ ln_x_g, const float* __restrict__ ln_x_b,
    const float* __restrict__ vn_w,
    const unsigned short* __restrict__ wbf,
    const float* __restrict__ bq, const float* __restrict__ bk, const float* __restrict__ bv,
    const float* __restrict__ bvd,
    float* __restrict__ q, float* __restrict__ k, float* __restrict__ v,
    float* __restrict__ vec_n, float* __restrict__ vec3w,
    float* __restrict__ vWtrg, float* __restrict__ vWsrc, float* __restrict__ vec_dot) {
    __shared__ __align__(16) unsigned short sA[8][16 * CC];

    const int lane = threadIdx.x & 31, wid = threadIdx.x >> 5;
    const int lm = lane & 15, lgrp = lane >> 4;
    const int tile = blockIdx.x * 8 + wid;
    if (tile >= NODE_TILES) return;
    const int n0 = tile * 16;
    unsigned short* tA = &sA[wid][0];

    // ---- layernorm(x) -> bf16 LDS tile ----
    {
        float4 g4 = ((const float4*)ln_x_g)[lane];
        float4 b4 = ((const float4*)ln_x_b)[lane];
#pragma unroll 4
        for (int m = 0; m < 16; ++m) {
            float4 val = ((const float4*)(x + (size_t)(n0 + m) * CC))[lane];
            float s  = wavesum(val.x + val.y + val.z + val.w);
            float s2 = wavesum(val.x * val.x + val.y * val.y + val.z * val.z + val.w * val.w);
            float mu = s * (1.f / 128.f);
            float rstd = rsqrtf(s2 * (1.f / 128.f) - mu * mu + 1e-5f);
            int c = lane * 4;
            st2bf(tA + m * CC + c,     (val.x - mu) * rstd * g4.x + b4.x,
                                       (val.y - mu) * rstd * g4.y + b4.y);
            st2bf(tA + m * CC + c + 2, (val.z - mu) * rstd * g4.z + b4.z,
                                       (val.w - mu) * rstd * g4.w + b4.w);
        }
    }
    // ---- q/k/v projections ----
    {
        v8f acc[8];
        const unsigned short* Wp[3] = {wbf + OWQ, wbf + OWK, wbf + OWV};
        const float* Bp[3] = {bq, bk, bv};
        float* Op[3] = {q, k, v};
#pragma unroll
        for (int p = 0; p < 3; ++p) {
#pragma unroll
            for (int nt = 0; nt < 8; ++nt) acc[nt] = zero8();
            gemmL<8>(tA, Wp[p], acc, lm, lgrp);
#pragma unroll
            for (int nt = 0; nt < 8; ++nt) {
                int c = nt * 16 + lm;
                float bias = Bp[p][c];
#pragma unroll
                for (int r = 0; r < 8; ++r)
                    Op[p][(size_t)(n0 + r + 8 * lgrp) * CC + c] = acc[nt][r] + bias;
            }
        }
    }
    // ---- vec_layernorm (max_min) -> vec_n (global, fp32) ----
    {
        float4 w4 = ((const float4*)vn_w)[lane];
#pragma unroll 2
        for (int m = 0; m < 16; ++m) {
            const float* vr = vec + (size_t)(n0 + m) * 3 * CC;
            float4 a0 = ((const float4*)(vr))[lane];
            float4 a1 = ((const float4*)(vr + CC))[lane];
            float4 a2 = ((const float4*)(vr + 2 * CC))[lane];
            float d0 = fmaxf(sqrtf(a0.x * a0.x + a1.x * a1.x + a2.x * a2.x), 1e-12f);
            float d1 = fmaxf(sqrtf(a0.y * a0.y + a1.y * a1.y + a2.y * a2.y), 1e-12f);
            float d2 = fmaxf(sqrtf(a0.z * a0.z + a1.z * a1.z + a2.z * a2.z), 1e-12f);
            float d3 = fmaxf(sqrtf(a0.w * a0.w + a1.w * a1.w + a2.w * a2.w), 1e-12f);
            float mn = fminf(fminf(d0, d1), fminf(d2, d3));
            float mx = fmaxf(fmaxf(d0, d1), fmaxf(d2, d3));
#pragma unroll
            for (int msk = 16; msk > 0; msk >>= 1) {
                mn = fminf(mn, __shfl_xor(mn, msk, 32));
                mx = fmaxf(mx, __shfl_xor(mx, msk, 32));
            }
            float delta = mx - mn;
            delta = (delta == 0.f) ? 1.f : delta;
            float rdelta = 1.f / delta;
            float f0 = fmaxf((d0 - mn) * rdelta, 0.f) / d0 * w4.x;
            float f1 = fmaxf((d1 - mn) * rdelta, 0.f) / d1 * w4.y;
            float f2 = fmaxf((d2 - mn) * rdelta, 0.f) / d2 * w4.z;
            float f3 = fmaxf((d3 - mn) * rdelta, 0.f) / d3 * w4.w;
            float* vo = vec_n + (size_t)(n0 + m) * 3 * CC;
            float4 o;
            o.x = a0.x * f0; o.y = a0.y * f1; o.z = a0.z * f2; o.w = a0.w * f3;
            ((float4*)(vo))[lane] = o;
            o.x = a1.x * f0; o.y = a1.y * f1; o.z = a1.z * f2; o.w = a1.w * f3;
            ((float4*)(vo + CC))[lane] = o;
            o.x = a2.x * f0; o.y = a2.y * f1; o.z = a2.z * f2; o.w = a2.w * f3;
            ((float4*)(vo + 2 * CC))[lane] = o;
        }
    }
    // ---- per-axis GEMMs: Wvec (vec1*vec2 accum, vec3), Wtrg, Wsrc ----
    v8f vdacc[8];
#pragma unroll
    for (int nt = 0; nt < 8; ++nt) vdacc[nt] = zero8();

    for (int axis = 0; axis < 3; ++axis) {
#pragma unroll 4
        for (int m = 0; m < 16; ++m) {
            float4 t = ((const float4*)(vec_n + ((size_t)(n0 + m) * 3 + axis) * CC))[lane];
            int c = lane * 4;
            st2bf(tA + m * CC + c,     t.x, t.y);
            st2bf(tA + m * CC + c + 2, t.z, t.w);
        }
        {   // vec1 and vec2 (two 8-tile GEMMs), accumulate elementwise product
            v8f acc1[8], acc2[8];
#pragma unroll
            for (int nt = 0; nt < 8; ++nt) { acc1[nt] = zero8(); acc2[nt] = zero8(); }
            gemmL<8>(tA, wbf + OWVEC, acc1, lm, lgrp);
            gemmL<8>(tA, wbf + OWVEC + 16384, acc2, lm, lgrp);
#pragma unroll
            for (int nt = 0; nt < 8; ++nt)
#pragma unroll
                for (int r = 0; r < 8; ++r)
                    vdacc[nt][r] += acc1[nt][r] * acc2[nt][r];
        }
        {   // vec3
            v8f acc[8];
#pragma unroll
            for (int nt = 0; nt < 8; ++nt) acc[nt] = zero8();
            gemmL<8>(tA, wbf + OWVEC + 32768, acc, lm, lgrp);
#pragma unroll
            for (int nt = 0; nt < 8; ++nt) {
                int c = nt * 16 + lm;
#pragma unroll
                for (int r = 0; r < 8; ++r)
                    vec3w[((size_t)(n0 + r + 8 * lgrp) * 3 + axis) * CC + c] = acc[nt][r];
            }
        }
        {   // Wtrg / Wsrc
            v8f acct[8], accs[8];
#pragma unroll
            for (int nt = 0; nt < 8; ++nt) { acct[nt] = zero8(); accs[nt] = zero8(); }
            gemmL<8>(tA, wbf + OWTRG, acct, lm, lgrp);
            gemmL<8>(tA, wbf + OWSRC, accs, lm, lgrp);
#pragma unroll
            for (int nt = 0; nt < 8; ++nt) {
                int c = nt * 16 + lm;
#pragma unroll
                for (int r = 0; r < 8; ++r) {
                    size_t idx = ((size_t)(n0 + r + 8 * lgrp) * 3 + axis) * CC + c;
                    vWtrg[idx] = acct[nt][r];
                    vWsrc[idx] = accs[nt][r];
                }
            }
        }
    }
    // ---- vec_dot = silu((vec1*vec2).sum(axis) @ Wvd^T + bvd) ----
    {
#pragma unroll
        for (int nt = 0; nt < 8; ++nt) {
            int c = nt * 16 + lm;
#pragma unroll
            for (int r = 0; r < 8; ++r)
                tA[(r + 8 * lgrp) * CC + c] = f2bf(vdacc[nt][r]);
        }
        v8f acc[8];
#pragma unroll
        for (int nt = 0; nt < 8; ++nt) acc[nt] = zero8();
        gemmL<8>(tA, wbf + OWVD, acc, lm, lgrp);
#pragma unroll
        for (int nt = 0; nt < 8; ++nt) {
            int c = nt * 16 + lm;
            float bias = bvd[c];
#pragma unroll
            for (int r = 0; r < 8; ++r)
                vec_dot[(size_t)(n0 + r + 8 * lgrp) * CC + c] = siluf(acc[nt][r] + bias);
        }
    }
}

// ---------------- fused edge stage ----------------
__global__ __launch_bounds__(256, 4) void edge_stage(
    const float* __restrict__ f_ij, const float* __restrict__ r_ij, const float* __restrict__ d_ij,
    const float* __restrict__ ln_f_g, const float* __restrict__ ln_f_b,
    const unsigned short* __restrict__ wbf,
    const float* __restrict__ bdk, const float* __restrict__ bdv,
    const float* __restrict__ bs, const float* __restrict__ bf_, const float* __restrict__ bwd,
    const int* __restrict__ edge_index,
    const float* __restrict__ q, const float* __restrict__ k, const float* __restrict__ v,
    const float* __restrict__ vec_n, const float* __restrict__ vWtrg, const float* __restrict__ vWsrc,
    float* __restrict__ x_agg, float* __restrict__ vec_agg, float* __restrict__ df_out) {
    __shared__ __align__(16) unsigned short sF[8][16 * CC];
    __shared__ __align__(16) unsigned short sT[8][16 * CC];

    const int lane = threadIdx.x & 31, wid = threadIdx.x >> 5;
    const int lm = lane & 15, lgrp = lane >> 4;
    const int tile = blockIdx.x * 8 + wid;
    if (tile >= EDGE_TILES) return;
    const int e0 = tile * 16;
    unsigned short* tF = &sF[wid][0];
    unsigned short* tT = &sT[wid][0];

    // prefetch next tile's f rows (hint only)
    __builtin_prefetch(f_ij + (size_t)(e0 + 16) * CC + lane * 16, 0, 1);

    // ---- per-edge metadata (edge e = e0 + r + 8*lgrp, matching D-tile rows) ----
    int srcN[8], dstN[8];
    float cut[8], dij0[8], dij1[8], dij2[8];
#pragma unroll
    for (int r = 0; r < 8; ++r) {
        int e = e0 + r + 8 * lgrp;
        srcN[r] = edge_index[e];
        dstN[r] = edge_index[N_EDGES + e];
        float rr = r_ij[e];
        cut[r] = 0.5f * (__cosf(rr * (PI_F / 5.f)) + 1.f) * ((rr < 5.f) ? 1.f : 0.f);
        dij0[r] = d_ij[(size_t)e * 3 + 0];
        dij1[r] = d_ij[(size_t)e * 3 + 1];
        dij2[r] = d_ij[(size_t)e * 3 + 2];
    }

    // ---- layernorm(f_ij) -> bf16 LDS ----
    {
        float4 g4 = ((const float4*)ln_f_g)[lane];
        float4 b4 = ((const float4*)ln_f_b)[lane];
#pragma unroll 4
        for (int m = 0; m < 16; ++m) {
            float4 val = ((const float4*)(f_ij + (size_t)(e0 + m) * CC))[lane];
            float s  = wavesum(val.x + val.y + val.z + val.w);
            float s2 = wavesum(val.x * val.x + val.y * val.y + val.z * val.z + val.w * val.w);
            float mu = s * (1.f / 128.f);
            float rstd = rsqrtf(s2 * (1.f / 128.f) - mu * mu + 1e-5f);
            int c = lane * 4;
            st2bf(tF + m * CC + c,     (val.x - mu) * rstd * g4.x + b4.x,
                                       (val.y - mu) * rstd * g4.y + b4.y);
            st2bf(tF + m * CC + c + 2, (val.z - mu) * rstd * g4.z + b4.z,
                                       (val.w - mu) * rstd * g4.w + b4.w);
        }
    }
    AF Af; loadAF(tF, Af, lm, lgrp);   // reused for dk, dv, Wf

    // ---- dk GEMM + attention logits (head h == n-tile, D=16 == tile width) ----
    float attn[8][8];
    {
        v8f acc[8];
#pragma unroll
        for (int nt = 0; nt < 8; ++nt) acc[nt] = zero8();
        gemmR<8>(Af, wbf + OWDK, acc, lm, lgrp);
#pragma unroll
        for (int nt = 0; nt < 8; ++nt) {
            int c = nt * 16 + lm;
            float bias = bdk[c];
#pragma unroll
            for (int r = 0; r < 8; ++r) {
                float dkv = siluf(acc[nt][r] + bias);
                float t = q[(size_t)dstN[r] * CC + c] * k[(size_t)srcN[r] * CC + c] * dkv;
                t = rowsum16(t);
                attn[r][nt] = siluf(t) * cut[r];
            }
        }
    }
    // ---- dv GEMM, vj, scatter x_agg, stage vj for Ws ----
    {
        v8f acc[8];
#pragma unroll
        for (int nt = 0; nt < 8; ++nt) acc[nt] = zero8();
        gemmR<8>(Af, wbf + OWDV, acc, lm, lgrp);
#pragma unroll
        for (int nt = 0; nt < 8; ++nt) {
            int c = nt * 16 + lm;
            float bias = bdv[c];
#pragma unroll
            for (int r = 0; r < 8; ++r) {
                float dvv = siluf(acc[nt][r] + bias);
                float vj = v[(size_t)srcN[r] * CC + c] * dvv * attn[r][nt];
                __hip_atomic_fetch_add(&x_agg[(size_t)dstN[r] * CC + c], vj,
                                       __ATOMIC_RELAXED, __HIP_MEMORY_SCOPE_AGENT);
                tT[(r + 8 * lgrp) * CC + c] = f2bf(vj);
            }
        }
    }
    // ---- Ws GEMM -> s1,s2 ; vec_msg scatter into vec_agg ----
    {
        v8f acc[16];
#pragma unroll
        for (int nt = 0; nt < 16; ++nt) acc[nt] = zero8();
        gemmL<16>(tT, wbf + OWS, acc, lm, lgrp);
#pragma unroll
        for (int nt = 0; nt < 8; ++nt) {
            int c = nt * 16 + lm;
            float b1 = bs[c], b2 = bs[CC + c];
#pragma unroll
            for (int r = 0; r < 8; ++r) {
                float s1 = siluf(acc[nt][r] + b1);
                float s2 = siluf(acc[8 + nt][r] + b2);
                size_t sb = (size_t)srcN[r] * 3 * CC + c;
                size_t db = (size_t)dstN[r] * 3 * CC + c;
                __hip_atomic_fetch_add(&vec_agg[db], vec_n[sb] * s1 + s2 * dij0[r],
                                       __ATOMIC_RELAXED, __HIP_MEMORY_SCOPE_AGENT);
                __hip_atomic_fetch_add(&vec_agg[db + CC], vec_n[sb + CC] * s1 + s2 * dij1[r],
                                       __ATOMIC_RELAXED, __HIP_MEMORY_SCOPE_AGENT);
                __hip_atomic_fetch_add(&vec_agg[db + 2 * CC], vec_n[sb + 2 * CC] * s1 + s2 * dij2[r],
                                       __ATOMIC_RELAXED, __HIP_MEMORY_SCOPE_AGENT);
            }
        }
    }
    // ---- edge update: wdot_in = <g1,g2> - <g1,d><g2,d> (rejection identity) ----
    {
#pragma unroll
        for (int nt = 0; nt < 8; ++nt) {
            int c = nt * 16 + lm;
#pragma unroll
            for (int r = 0; r < 8; ++r) {
                size_t tb = (size_t)dstN[r] * 3 * CC + c;
                size_t sb = (size_t)srcN[r] * 3 * CC + c;
                float g10 = vWtrg[tb], g11 = vWtrg[tb + CC], g12 = vWtrg[tb + 2 * CC];
                float g20 = vWsrc[sb], g21 = vWsrc[sb + CC], g22 = vWsrc[sb + 2 * CC];
                float s12 = g10 * g20 + g11 * g21 + g12 * g22;
                float p1 = g10 * dij0[r] + g11 * dij1[r] + g12 * dij2[r];
                float p2 = g20 * dij0[r] + g21 * dij1[r] + g22 * dij2[r];
                tT[(r + 8 * lgrp) * CC + c] = f2bf(s12 - p1 * p2);
            }
        }
    }
    // ---- Wwd GEMM -> w_dot (silu) stashed back into tT as bf16 (frees regs) ----
    {
        v8f accw[8];
#pragma unroll
        for (int nt = 0; nt < 8; ++nt) accw[nt] = zero8();
        gemmL<8>(tT, wbf + OWWD, accw, lm, lgrp);
#pragma unroll
        for (int nt = 0; nt < 8; ++nt) {
            int c = nt * 16 + lm;
            float bw = bwd[c];
#pragma unroll
            for (int r = 0; r < 8; ++r)
                tT[(r + 8 * lgrp) * CC + c] = f2bf(siluf(accw[nt][r] + bw));
        }
    }
    // ---- Wf GEMM (f1,f2) -> df = f1*w_dot + f2 ----
    {
        v8f accf[16];
#pragma unroll
        for (int nt = 0; nt < 16; ++nt) accf[nt] = zero8();
        gemmR<16>(Af, wbf + OWF, accf, lm, lgrp);
#pragma unroll
        for (int nt = 0; nt < 8; ++nt) {
            int c = nt * 16 + lm;
            float b1 = bf_[c], b2 = bf_[CC + c];
#pragma unroll
            for (int r = 0; r < 8; ++r) {
                float wd = bf2f(tT[(r + 8 * lgrp) * CC + c]);
                float f1 = siluf(accf[nt][r] + b1);
                float f2 = siluf(accf[8 + nt][r] + b2);
                df_out[(size_t)(e0 + r + 8 * lgrp) * CC + c] = f1 * wd + f2;
            }
        }
    }
}

// ---------------- node finalize ----------------
__global__ __launch_bounds__(256, 4) void node_out(
    const float* __restrict__ x_agg, const float* __restrict__ vec_agg,
    const float* __restrict__ vec3w, const float* __restrict__ vec_dot,
    const unsigned short* __restrict__ wbf, const float* __restrict__ bo,
    float* __restrict__ dx_out, float* __restrict__ dvec_out) {
    __shared__ __align__(16) unsigned short sA[8][16 * CC];

    const int lane = threadIdx.x & 31, wid = threadIdx.x >> 5;
    const int lm = lane & 15, lgrp = lane >> 4;
    const int tile = blockIdx.x * 8 + wid;
    if (tile >= NODE_TILES) return;
    const int n0 = tile * 16;
    unsigned short* tA = &sA[wid][0];

#pragma unroll 4
    for (int m = 0; m < 16; ++m) {
        float4 t = ((const float4*)(x_agg + (size_t)(n0 + m) * CC))[lane];
        int c = lane * 4;
        st2bf(tA + m * CC + c,     t.x, t.y);
        st2bf(tA + m * CC + c + 2, t.z, t.w);
    }
    AF Aa; loadAF(tA, Aa, lm, lgrp);

    // o2, o3 first -> dx; then o1 -> dvec (caps live accumulators at 128 VGPRs)
    {
        v8f acc2[8], acc3[8];
#pragma unroll
        for (int nt = 0; nt < 8; ++nt) { acc2[nt] = zero8(); acc3[nt] = zero8(); }
        gemmR<8>(Aa, wbf + OWO + 16384, acc2, lm, lgrp);
        gemmR<8>(Aa, wbf + OWO + 32768, acc3, lm, lgrp);
#pragma unroll
        for (int nt = 0; nt < 8; ++nt) {
            int c = nt * 16 + lm;
            float b2 = bo[CC + c], b3 = bo[2 * CC + c];
#pragma unroll
            for (int r = 0; r < 8; ++r) {
                int n = n0 + r + 8 * lgrp;
                dx_out[(size_t)n * CC + c] =
                    vec_dot[(size_t)n * CC + c] * (acc2[nt][r] + b2) + (acc3[nt][r] + b3);
            }
        }
    }
    {
        v8f acc1[8];
#pragma unroll
        for (int nt = 0; nt < 8; ++nt) acc1[nt] = zero8();
        gemmR<8>(Aa, wbf + OWO, acc1, lm, lgrp);
#pragma unroll
        for (int nt = 0; nt < 8; ++nt) {
            int c = nt * 16 + lm;
            float b1 = bo[c];
#pragma unroll
            for (int r = 0; r < 8; ++r) {
                int n = n0 + r + 8 * lgrp;
                float o1 = acc1[nt][r] + b1;
                size_t vb = (size_t)n * 3 * CC + c;
                dvec_out[vb]          = vec3w[vb] * o1 + vec_agg[vb];
                dvec_out[vb + CC]     = vec3w[vb + CC] * o1 + vec_agg[vb + CC];
                dvec_out[vb + 2 * CC] = vec3w[vb + 2 * CC] * o1 + vec_agg[vb + 2 * CC];
            }
        }
    }
}

// ---------------- host side ----------------
static inline void cvt(const float* s, unsigned short* d, int n, hipStream_t st) {
    cvt_bf16_kernel<<<(n + 255) / 256, 256, 0, st>>>(s, d, n);
}

extern "C" void kernel_launch(void* const* d_in, const int* in_sizes, int n_in,
                              void* d_out, int out_size, void* d_ws, size_t ws_size,
                              hipStream_t stream) {
    (void)in_sizes; (void)n_in; (void)out_size; (void)ws_size;
    const float* x      = (const float*)d_in[0];
    const float* vec    = (const float*)d_in[1];
    const float* r_ij   = (const float*)d_in[2];
    const float* f_ij   = (const float*)d_in[3];
    const float* d_ij   = (const float*)d_in[4];
    const float* ln_x_g = (const float*)d_in[5];
    const float* ln_x_b = (const float*)d_in[6];
    const float* ln_f_g = (const float*)d_in[7];
    const float* ln_f_b = (const float*)d_in[8];
    const float* vn_w   = (const float*)d_in[9];
    const float* Wq = (const float*)d_in[10]; const float* bq  = (const float*)d_in[11];
    const float* Wk = (const float*)d_in[12]; const float* bk  = (const float*)d_in[13];
    const float* Wv = (const float*)d_in[14]; const float* bv  = (const float*)d_in[15];
    const float* Wvd = (const float*)d_in[16]; const float* bvd = (const float*)d_in[17];
    const float* Ws = (const float*)d_in[18]; const float* bs  = (const float*)d_in[19];
    const float* Wo = (const float*)d_in[20]; const float* bo  = (const float*)d_in[21];
    const float* Wvec = (const float*)d_in[22];
    const float* Wdk = (const float*)d_in[23]; const float* bdk = (const float*)d_in[24];
    const float* Wdv = (const float*)d_in[25]; const float* bdv = (const float*)d_in[26];
    const float* Wf = (const float*)d_in[27]; const float* bf_ = (const float*)d_in[28];
    const float* Wsrc = (const float*)d_in[29];
    const float* Wtrg = (const float*)d_in[30];
    const float* Wwd = (const float*)d_in[31]; const float* bwd = (const float*)d_in[32];
    const int* edge_index = (const int*)d_in[33];

    unsigned short* wbf = (unsigned short*)d_ws;
    float* fws = (float*)((char*)d_ws + (size_t)WBF_TOTAL * 2);
    float* q       = fws;
    float* k       = q + NC;
    float* v       = k + NC;
    float* vec_n   = v + NC;
    float* vec3w   = vec_n + N3C;
    float* vWtrg   = vec3w + N3C;
    float* vWsrc   = vWtrg + N3C;
    float* vec_dot = vWsrc + N3C;
    float* x_agg   = vec_dot + NC;
    float* vec_agg = x_agg + NC;   // x_agg, vec_agg contiguous (zeroed together)

    float* dx_out   = (float*)d_out;
    float* dvec_out = dx_out + NC;
    float* df_out   = dvec_out + N3C;

    // weights -> bf16
    cvt(Wq, wbf + OWQ, 16384, stream);
    cvt(Wk, wbf + OWK, 16384, stream);
    cvt(Wv, wbf + OWV, 16384, stream);
    cvt(Wvec, wbf + OWVEC, 49152, stream);
    cvt(Wtrg, wbf + OWTRG, 16384, stream);
    cvt(Wsrc, wbf + OWSRC, 16384, stream);
    cvt(Wvd, wbf + OWVD, 16384, stream);
    cvt(Wdk, wbf + OWDK, 16384, stream);
    cvt(Wdv, wbf + OWDV, 16384, stream);
    cvt(Ws, wbf + OWS, 32768, stream);
    cvt(Wf, wbf + OWF, 32768, stream);
    cvt(Wwd, wbf + OWWD, 16384, stream);
    cvt(Wo, wbf + OWO, 49152, stream);

    // zero aggregation buffers
    zerof_kernel<<<(NC + N3C + 255) / 256, 256, 0, stream>>>(x_agg, NC + N3C);

    node_stage<<<(NODE_TILES + 7) / 8, 256, 0, stream>>>(
        x, vec, ln_x_g, ln_x_b, vn_w, wbf, bq, bk, bv, bvd,
        q, k, v, vec_n, vec3w, vWtrg, vWsrc, vec_dot);

    edge_stage<<<EDGE_TILES / 8, 256, 0, stream>>>(
        f_ij, r_ij, d_ij, ln_f_g, ln_f_b, wbf, bdk, bdv, bs, bf_, bwd,
        edge_index, q, k, v, vec_n, vWtrg, vWsrc, x_agg, vec_agg, df_out);

    node_out<<<(NODE_TILES + 7) / 8, 256, 0, stream>>>(
        x_agg, vec_agg, vec3w, vec_dot, wbf, bo, dx_out, dvec_out);
}
